// Transformer_45114336477514
// MI455X (gfx1250) — matrix operand order repaired and checked by reading, not compiled, not run on hardware
//
#include <hip/hip_runtime.h>
#include <cstddef>

#define BATCH 16
#define S_LEN 1024
#define IN_D  196
#define K1P   224   // 196 padded to 7*32
#define MDL   768
#define NH    8
#define DK    96
#define N1P   224   // 196 padded to 7*32 (N must be multiple of 32 for 2x2 blocking)
#define M_TOK (BATCH * S_LEN)

typedef __attribute__((ext_vector_type(16))) _Float16 v16h;
typedef __attribute__((ext_vector_type(8)))  _Float16 v8h;
typedef __attribute__((ext_vector_type(8)))  float    v8f;

__device__ __forceinline__ v16h frag_combine(v8h lo, v8h hi) {
  v16h r;
#pragma unroll
  for (int i = 0; i < 8; ++i) { r[i] = lo[i]; r[i + 8] = hi[i]; }
  return r;
}

// A-operand fragment (16x32, 16-bit): lanes 0-15 hold M=lane, K {0..7,16..23};
// lanes 16-31 hold M=lane-16, K {8..15,24..31}.  rowk = &A[row][kbase]
__device__ __forceinline__ v16h load_afrag(const _Float16* rowk, int lane) {
  int koff = (lane & 16) >> 1;                 // 0 or 8
  v8h lo = *(const v8h*)(rowk + koff);
  v8h hi = *(const v8h*)(rowk + koff + 16);
  return frag_combine(lo, hi);
}

// B-operand fragment (32x16, 16-bit): lanes 0-15 hold N=lane, K {0..7,16..23};
// lanes 16-31 hold N=lane-16, K {8..15,24..31}.  rowk = &B[ncol][kbase]
__device__ __forceinline__ v16h load_bfrag(const _Float16* rowk, int lane) {
  int koff = (lane & 16) >> 1;                 // 0 or 8
  v8h lo = *(const v8h*)(rowk + koff);
  v8h hi = *(const v8h*)(rowk + koff + 16);
  return frag_combine(lo, hi);
}

#define WMMA_F16(a, b, c) \
  __builtin_amdgcn_wmma_f32_16x16x32_f16(false, (a), false, (b), (short)0, (c), false, false)

// ---------------------------------------------------------------------------
// Weight convert + pad: f32 [N,Kin] -> f16 [Npad,Kpad] (zero padded)
// ---------------------------------------------------------------------------
__global__ void k_cvt_w(const float* __restrict__ src, _Float16* __restrict__ dst,
                        int N, int Kin, int Npad, int Kpad) {
  int idx = blockIdx.x * 256 + threadIdx.x;
  if (idx >= Npad * Kpad) return;
  int r = idx / Kpad, c = idx - r * Kpad;
  dst[idx] = (r < N && c < Kin) ? (_Float16)src[r * Kin + c] : (_Float16)0.f;
}

// ---------------------------------------------------------------------------
// LayerNorm over 196 dims. One wave per token (4 waves / 128-thread block).
// mode A (out16 != nullptr): write f16 padded [token, K1P]
// mode B (out16 == nullptr): outf32[token,196] = res[token,196] + LN(x)
// ---------------------------------------------------------------------------
__global__ void k_ln(const float* __restrict__ x, const float* __restrict__ w,
                     const float* __restrict__ b, _Float16* __restrict__ out16,
                     float* __restrict__ outf32, const float* __restrict__ res) {
  int token = blockIdx.x * 4 + (threadIdx.x >> 5);
  int lane  = threadIdx.x & 31;
  const float* row = x + (size_t)token * IN_D;
  float s = 0.f, sq = 0.f;
  for (int i = lane; i < IN_D; i += 32) { float t = row[i]; s += t; sq += t * t; }
#pragma unroll
  for (int off = 16; off >= 1; off >>= 1) {
    s  += __shfl_xor(s, off, 32);
    sq += __shfl_xor(sq, off, 32);
  }
  float mean = s * (1.f / IN_D);
  float var  = sq * (1.f / IN_D) - mean * mean;
  float rstd = rsqrtf(var + 1e-5f);
  if (out16) {
    _Float16* o = out16 + (size_t)token * K1P;
    for (int i = lane; i < K1P; i += 32) {
      float v = (i < IN_D) ? (row[i] - mean) * rstd * w[i] + b[i] : 0.f;
      o[i] = (_Float16)v;
    }
  } else {
    float* o = outf32 + (size_t)token * IN_D;
    const float* r2 = res + (size_t)token * IN_D;
    for (int i = lane; i < IN_D; i += 32)
      o[i] = r2[i] + (row[i] - mean) * rstd * w[i] + b[i];
  }
}

// ---------------------------------------------------------------------------
// Generic WMMA GEMM, 32x32 tile per wave (2x2 WMMA micro-kernel):
//   C[m,n] = sum_k A[m,k] * B[n,k]  (+ bias[n])
// A: f16 [Mt32*32, Kp] row-major.  B: f16 [>=Nt32*32, Kp] row-major.
// 256-thread blocks = 8 waves = 8 tiles.
// mode 0: outf32[m*Nreal+n] = C+bias (+residual) for n < Nreal
// mode 2: outf16[m*ld+n] = gelu_exact(C+bias)   (ld = Nreal = 768)
// mode 3: scatter Q/K -> f16 [b,h,s,d]   (n = h*DK+d, m = b*S+s)
// mode 4: scatter V   -> f16 [b,h,d,s]
// ---------------------------------------------------------------------------
__global__ void k_gemm(const _Float16* __restrict__ A, const _Float16* __restrict__ B,
                       const float* __restrict__ bias, int Mt32, int Nt32, int Kp,
                       int Nreal, int mode, float* __restrict__ outf32,
                       const float* __restrict__ residual, _Float16* __restrict__ outf16) {
  int wave = (int)((blockIdx.x * blockDim.x + threadIdx.x) >> 5);
  int lane = threadIdx.x & 31;
  if (wave >= Mt32 * Nt32) return;
  int tm = (wave / Nt32) * 32;
  int tn = (wave % Nt32) * 32;
  int ln15 = lane & 15;
  const _Float16* arow0 = A + (size_t)(tm + ln15) * Kp;
  const _Float16* arow1 = arow0 + (size_t)16 * Kp;
  const _Float16* brow0 = B + (size_t)(tn + ln15) * Kp;
  const _Float16* brow1 = brow0 + (size_t)16 * Kp;
  v8f c00, c01, c10, c11;
#pragma unroll
  for (int i = 0; i < 8; ++i) { c00[i] = 0.f; c01[i] = 0.f; c10[i] = 0.f; c11[i] = 0.f; }
  for (int k = 0; k < Kp; k += 32) {
    v16h a0 = load_afrag(arow0 + k, lane);
    v16h a1 = load_afrag(arow1 + k, lane);
    v16h b0 = load_bfrag(brow0 + k, lane);
    v16h b1 = load_bfrag(brow1 + k, lane);
    c00 = WMMA_F16(a0, b0, c00);
    c01 = WMMA_F16(a0, b1, c01);
    c10 = WMMA_F16(a1, b0, c10);
    c11 = WMMA_F16(a1, b1, c11);
  }
  int rg = (lane >> 4) << 3;
#pragma unroll
  for (int mi = 0; mi < 2; ++mi) {
#pragma unroll
    for (int ni = 0; ni < 2; ++ni) {
      const v8f& c = (mi == 0) ? (ni == 0 ? c00 : c01) : (ni == 0 ? c10 : c11);
      int n = tn + ni * 16 + ln15;
      int rbase = tm + mi * 16 + rg;
      float bv = (n < Nreal) ? bias[n] : 0.f;
      if (mode == 0) {
        if (n < Nreal) {
#pragma unroll
          for (int i = 0; i < 8; ++i) {
            size_t idx = (size_t)(rbase + i) * Nreal + n;
            float v = c[i] + bv;
            if (residual) v += residual[idx];
            outf32[idx] = v;
          }
        }
      } else if (mode == 2) {
#pragma unroll
        for (int i = 0; i < 8; ++i) {
          float xg = c[i] + bv;
          float g = 0.5f * xg * (1.f + erff(xg * 0.70710678118654752f));
          outf16[(size_t)(rbase + i) * Nreal + n] = (_Float16)g;
        }
      } else if (mode == 3) {
        int hh = n / DK, d = n - hh * DK;
#pragma unroll
        for (int i = 0; i < 8; ++i) {
          int m = rbase + i;
          int bb = m >> 10, ss = m & (S_LEN - 1);
          outf16[(((size_t)(bb * NH + hh) * S_LEN) + ss) * DK + d] = (_Float16)(c[i] + bv);
        }
      } else {  // mode 4
        int hh = n / DK, d = n - hh * DK;
#pragma unroll
        for (int i = 0; i < 8; ++i) {
          int m = rbase + i;
          int bb = m >> 10, ss = m & (S_LEN - 1);
          outf16[(((size_t)(bb * NH + hh) * DK) + d) * S_LEN + ss] = (_Float16)(c[i] + bv);
        }
      }
    }
  }
}

// ---------------------------------------------------------------------------
// Flash attention. grid = (8, NH, BATCH), block = 256 (8 waves).
// Each wave: one 16-row Q tile, streams 32-key tiles with online softmax.
// q,k: f16 [b,h,s,DK]; v: f16 [b,h,DK,s] (transposed); out ctx16 f16 [tok, MDL]
// ---------------------------------------------------------------------------
__global__ void k_attn(const _Float16* __restrict__ qh, const _Float16* __restrict__ kh,
                       const _Float16* __restrict__ vt, _Float16* __restrict__ ctx16) {
  __shared__ _Float16 lds[8][16][32];
  int wv = threadIdx.x >> 5;
  int lane = threadIdx.x & 31;
  int ln15 = lane & 15;
  int qtile = blockIdx.x * 8 + wv;
  int head = blockIdx.y, batch = blockIdx.z;
  size_t bh = (size_t)batch * NH + head;
  const _Float16* qb = qh + bh * S_LEN * DK;
  const _Float16* kb = kh + bh * S_LEN * DK;
  const _Float16* vb = vt + bh * DK * S_LEN;

  int qrow = qtile * 16 + ln15;
  v16h aq[3];
#pragma unroll
  for (int cc = 0; cc < 3; ++cc)
    aq[cc] = load_afrag(qb + (size_t)qrow * DK + 32 * cc, lane);

  v8f acc[6];
#pragma unroll
  for (int cc = 0; cc < 6; ++cc)
#pragma unroll
    for (int i = 0; i < 8; ++i) acc[cc][i] = 0.f;
  float mstat[8], lsum[8];
#pragma unroll
  for (int i = 0; i < 8; ++i) { mstat[i] = -1e30f; lsum[i] = 0.f; }

  const float scl = 0.35355339059327f;  // 1/sqrt(NUM_HEADS)  (faithful quirk)
  int rg = (lane >> 4) << 3;

  for (int kt = 0; kt < S_LEN; kt += 32) {
    v8f s0, s1;
#pragma unroll
    for (int i = 0; i < 8; ++i) { s0[i] = 0.f; s1[i] = 0.f; }
    const _Float16* kr0 = kb + (size_t)(kt + ln15) * DK;
    const _Float16* kr1 = kb + (size_t)(kt + 16 + ln15) * DK;
#pragma unroll
    for (int cc = 0; cc < 3; ++cc) {
      v16h b0 = load_bfrag(kr0 + 32 * cc, lane);
      s0 = WMMA_F16(aq[cc], b0, s0);
      v16h b1 = load_bfrag(kr1 + 32 * cc, lane);
      s1 = WMMA_F16(aq[cc], b1, s1);
    }
    float pm[8];
#pragma unroll
    for (int i = 0; i < 8; ++i) {
      s0[i] *= scl; s1[i] *= scl;
      pm[i] = fmaxf(s0[i], s1[i]);
    }
#pragma unroll
    for (int m = 1; m <= 8; m <<= 1)
#pragma unroll
      for (int i = 0; i < 8; ++i) pm[i] = fmaxf(pm[i], __shfl_xor(pm[i], m, 32));
    float p0[8], p1[8], rs[8];
#pragma unroll
    for (int i = 0; i < 8; ++i) {
      float mn = fmaxf(mstat[i], pm[i]);
      float corr = __expf(mstat[i] - mn);
      p0[i] = __expf(s0[i] - mn);
      p1[i] = __expf(s1[i] - mn);
      rs[i] = p0[i] + p1[i];
      mstat[i] = mn;
      lsum[i] *= corr;
#pragma unroll
      for (int cc = 0; cc < 6; ++cc) acc[cc][i] *= corr;
    }
#pragma unroll
    for (int m = 1; m <= 8; m <<= 1)
#pragma unroll
      for (int i = 0; i < 8; ++i) rs[i] += __shfl_xor(rs[i], m, 32);
#pragma unroll
    for (int i = 0; i < 8; ++i) lsum[i] += rs[i];

    // Stage P (C-layout) into LDS, reload as A-operand layout.
#pragma unroll
    for (int i = 0; i < 8; ++i) {
      lds[wv][rg + i][ln15]      = (_Float16)p0[i];
      lds[wv][rg + i][16 + ln15] = (_Float16)p1[i];
    }
    __syncthreads();
    v16h ap = load_afrag(&lds[wv][ln15][0], lane);
#pragma unroll
    for (int cc = 0; cc < 6; ++cc) {
      v16h bvv = load_bfrag(vb + (size_t)(cc * 16 + ln15) * S_LEN + kt, lane);
      acc[cc] = WMMA_F16(ap, bvv, acc[cc]);
    }
    __syncthreads();
  }

  float inv[8];
#pragma unroll
  for (int i = 0; i < 8; ++i) inv[i] = 1.f / lsum[i];
#pragma unroll
  for (int cc = 0; cc < 6; ++cc)
#pragma unroll
    for (int i = 0; i < 8; ++i) {
      int row = qtile * 16 + rg + i;
      ctx16[((size_t)(batch * S_LEN + row)) * MDL + head * DK + cc * 16 + ln15] =
          (_Float16)(acc[cc][i] * inv[i]);
    }
}

// ---------------------------------------------------------------------------
extern "C" void kernel_launch(void* const* d_in, const int* in_sizes, int n_in,
                              void* d_out, int out_size, void* d_ws, size_t ws_size,
                              hipStream_t stream) {
  const float* x       = (const float*)d_in[0];
  const float* ln_a_w  = (const float*)d_in[1];
  const float* ln_a_b  = (const float*)d_in[2];
  const float* q_w     = (const float*)d_in[3];
  const float* q_b     = (const float*)d_in[4];
  const float* k_w     = (const float*)d_in[5];
  const float* k_b     = (const float*)d_in[6];
  const float* v_w     = (const float*)d_in[7];
  const float* v_b     = (const float*)d_in[8];
  const float* o_w     = (const float*)d_in[9];
  const float* o_b     = (const float*)d_in[10];
  const float* ln_f1_w = (const float*)d_in[11];
  const float* ln_f1_b = (const float*)d_in[12];
  const float* ff1_w   = (const float*)d_in[13];
  const float* ff1_b   = (const float*)d_in[14];
  const float* ff2_w   = (const float*)d_in[15];
  const float* ff2_b   = (const float*)d_in[16];
  const float* ln_f2_w = (const float*)d_in[17];
  const float* ln_f2_b = (const float*)d_in[18];

  char* ws = (char*)d_ws;
  size_t off = 0;
  auto nxt = [&](size_t bytes) {
    char* p = ws + off;
    off += (bytes + 255) & ~(size_t)255;
    return p;
  };
  _Float16* wq16  = (_Float16*)nxt((size_t)MDL * K1P * 2);
  _Float16* wk16  = (_Float16*)nxt((size_t)MDL * K1P * 2);
  _Float16* wv16  = (_Float16*)nxt((size_t)MDL * K1P * 2);
  _Float16* wff1  = (_Float16*)nxt((size_t)MDL * K1P * 2);
  _Float16* wo16  = (_Float16*)nxt((size_t)N1P * MDL * 2);
  _Float16* wff2  = (_Float16*)nxt((size_t)N1P * MDL * 2);
  _Float16* h16   = (_Float16*)nxt((size_t)M_TOK * K1P * 2);     // reused for LN(x2)
  _Float16* qh    = (_Float16*)nxt((size_t)M_TOK * DK * NH * 2); // reused for gelu act
  _Float16* kh    = (_Float16*)nxt((size_t)M_TOK * DK * NH * 2); // reused for y2 (f32)
  _Float16* vt    = (_Float16*)nxt((size_t)M_TOK * DK * NH * 2);
  _Float16* ctx16 = (_Float16*)nxt((size_t)M_TOK * MDL * 2);
  float*    x2    = (float*)   nxt((size_t)M_TOK * IN_D * 4);
  _Float16* g16   = qh;          // reuse: FF1 output (f16 [M_TOK, MDL])
  float*    y2    = (float*)kh;  // reuse: FF2 output (f32 [M_TOK, IN_D])

  // --- weight conversion ---
  {
    int t1 = MDL * K1P, t2 = N1P * MDL;
    k_cvt_w<<<(t1 + 255) / 256, 256, 0, stream>>>(q_w,   wq16, MDL, IN_D, MDL, K1P);
    k_cvt_w<<<(t1 + 255) / 256, 256, 0, stream>>>(k_w,   wk16, MDL, IN_D, MDL, K1P);
    k_cvt_w<<<(t1 + 255) / 256, 256, 0, stream>>>(v_w,   wv16, MDL, IN_D, MDL, K1P);
    k_cvt_w<<<(t1 + 255) / 256, 256, 0, stream>>>(ff1_w, wff1, MDL, IN_D, MDL, K1P);
    k_cvt_w<<<(t2 + 255) / 256, 256, 0, stream>>>(o_w,   wo16, IN_D, MDL, N1P, MDL);
    k_cvt_w<<<(t2 + 255) / 256, 256, 0, stream>>>(ff2_w, wff2, IN_D, MDL, N1P, MDL);
  }

  const int Mt32  = M_TOK / 32;              // 512
  const int NtQ32 = MDL / 32;                // 24
  const int Nt132 = N1P / 32;                // 7
  const int gQ = (Mt32 * NtQ32 + 7) / 8;     // 1536 blocks of 256
  const int g1 = (Mt32 * Nt132 + 7) / 8;     // 448

  // --- attention branch ---
  k_ln<<<M_TOK / 4, 128, 0, stream>>>(x, ln_a_w, ln_a_b, h16, nullptr, nullptr);
  k_gemm<<<gQ, 256, 0, stream>>>(h16, wq16, q_b, Mt32, NtQ32, K1P, MDL, 3, nullptr, nullptr, qh);
  k_gemm<<<gQ, 256, 0, stream>>>(h16, wk16, k_b, Mt32, NtQ32, K1P, MDL, 3, nullptr, nullptr, kh);
  k_gemm<<<gQ, 256, 0, stream>>>(h16, wv16, v_b, Mt32, NtQ32, K1P, MDL, 4, nullptr, nullptr, vt);
  k_attn<<<dim3(S_LEN / 16 / 8, NH, BATCH), 256, 0, stream>>>(qh, kh, vt, ctx16);
  // x2 = x + ctx @ o_w.T + o_b
  k_gemm<<<g1, 256, 0, stream>>>(ctx16, wo16, o_b, Mt32, Nt132, MDL, IN_D, 0, x2, x, nullptr);

  // --- feed-forward branch ---
  k_ln<<<M_TOK / 4, 128, 0, stream>>>(x2, ln_f1_w, ln_f1_b, h16, nullptr, nullptr);
  k_gemm<<<gQ, 256, 0, stream>>>(h16, wff1, ff1_b, Mt32, NtQ32, K1P, MDL, 2, nullptr, nullptr, g16);
  k_gemm<<<g1, 256, 0, stream>>>(g16, wff2, ff2_b, Mt32, Nt132, MDL, IN_D, 0, y2, nullptr, nullptr);
  // out = x2 + LN(y2)
  k_ln<<<M_TOK / 4, 128, 0, stream>>>(y2, ln_f2_w, ln_f2_b, nullptr, (float*)d_out, x2);
}
